// DANetAttention_87960930222507
// MI455X (gfx1250) — compile-verified, matrix operand order checked by reading, not run on hardware
//
#include <hip/hip_runtime.h>
#include <math.h>

typedef __attribute__((ext_vector_type(16))) __bf16 v16bf;
typedef __attribute__((ext_vector_type(8)))  float  v8f;

static constexpr int BATCH = 16;
static constexpr int CIN   = 512;
static constexpr int COUTC = 512;
static constexpr int HH = 48, WWI = 48;
static constexpr int NSP = HH * WWI;      // 2304 spatial positions
static constexpr int INTER = 128;
static constexpr float BN_EPS = 1e-5f;

__device__ __forceinline__ int kfrag(int j, int hv) {
    // K index inside a 16x32 bf16 A-frag / 32x16 B-frag for element j of this lane
    return ((j >> 3) << 4) + (hv << 3) + (j & 7);
}

__device__ __forceinline__ v8f wmma_bf16(v16bf a, v16bf b, v8f c) {
    return __builtin_amdgcn_wmma_f32_16x16x32_bf16(false, a, false, b, (short)0, c, false, false);
}

// Load a 16-element bf16 fragment whose K-indices are the two contiguous
// 8-float runs  [hv*8, hv*8+8)  and  [16+hv*8, 16+hv*8+8)  of `row`.
// Emits 4x b128 loads (global or LDS depending on pointer provenance).
__device__ __forceinline__ v16bf load_frag_row(const float* row, int hv) {
    const float4* p0 = reinterpret_cast<const float4*>(row + hv * 8);
    const float4* p1 = reinterpret_cast<const float4*>(row + 16 + hv * 8);
    float4 x0 = p0[0], x1 = p0[1], y0 = p1[0], y1 = p1[1];
    v16bf r;
    r[0]  = (__bf16)x0.x; r[1]  = (__bf16)x0.y; r[2]  = (__bf16)x0.z; r[3]  = (__bf16)x0.w;
    r[4]  = (__bf16)x1.x; r[5]  = (__bf16)x1.y; r[6]  = (__bf16)x1.z; r[7]  = (__bf16)x1.w;
    r[8]  = (__bf16)y0.x; r[9]  = (__bf16)y0.y; r[10] = (__bf16)y0.z; r[11] = (__bf16)y0.w;
    r[12] = (__bf16)y1.x; r[13] = (__bf16)y1.y; r[14] = (__bf16)y1.z; r[15] = (__bf16)y1.w;
    return r;
}

// ---------------------------------------------------------------------------
// BN folding: y = a*x + b  with a = scale*rsqrt(var+eps), b = bias - mean*a
// ---------------------------------------------------------------------------
__global__ void k_bnprep(const float* __restrict__ scale, const float* __restrict__ bias,
                         const float* __restrict__ mean, const float* __restrict__ var,
                         float* __restrict__ bnA, float* __restrict__ bnB, int C) {
    int c = blockIdx.x * blockDim.x + threadIdx.x;
    if (c < C) {
        float a = scale[c] * rsqrtf(var[c] + BN_EPS);
        bnA[c] = a;
        bnB[c] = bias[c] - mean[c] * a;
    }
}

// ---------------------------------------------------------------------------
// Depthwise 3x3, pad=1, no bias. x,y layout: [B][C][H][W]
// ---------------------------------------------------------------------------
__global__ void k_dw3x3(const float* __restrict__ x, const float* __restrict__ w,
                        float* __restrict__ y, int C) {
    int idx = blockIdx.x * blockDim.x + threadIdx.x;
    int total = BATCH * C * NSP;
    if (idx >= total) return;
    int n = idx % NSP;
    int c = (idx / NSP) % C;
    int h = n / WWI, wc = n % WWI;
    const float* xp = x + (size_t)(idx - n);   // base of this [b][c] plane
    const float* wp = w + c * 9;
    float acc = 0.f;
#pragma unroll
    for (int kh = 0; kh < 3; ++kh) {
        int hh = h + kh - 1;
        if (hh < 0 || hh >= HH) continue;
#pragma unroll
        for (int kw = 0; kw < 3; ++kw) {
            int wcol = wc + kw - 1;
            if (wcol < 0 || wcol >= WWI) continue;
            acc += xp[hh * WWI + wcol] * wp[kh * 3 + kw];
        }
    }
    y[idx] = acc;
}

// ---------------------------------------------------------------------------
// Pointwise 1x1 conv as GEMM: out[b,co,n] = sum_ci W[co,ci] * act'[b,ci,n]
// act' = (act (+act2)) * dwS + dwB   (fused depthwise-1x1 / residual sum)
// Feature set resolved at compile time => straight-line fragment loaders.
// ---------------------------------------------------------------------------
template<int CIN_T, bool HAS_ACT2, bool HAS_DW, bool HAS_PWB, bool HAS_BN, bool RELU>
__global__ void __launch_bounds__(32)
k_pwgemm(const float* __restrict__ act, const float* __restrict__ act2,
         const float* __restrict__ Wt,  const float* __restrict__ dwS,
         const float* __restrict__ dwB, const float* __restrict__ pwb,
         const float* __restrict__ bnA, const float* __restrict__ bnB,
         float* __restrict__ out, int Cout) {
    int b = blockIdx.x, cot = blockIdx.y, nb = blockIdx.z;
    int lane = threadIdx.x & 31;
    int hv = lane >> 4, rc = lane & 15;
    int co0 = cot * 16, n0 = nb * 16;

    const float* wrow = Wt + (size_t)(co0 + rc) * CIN_T;          // A: weight row
    const float* ap   = act + (size_t)b * CIN_T * NSP + n0 + rc;  // B: channel-strided
    const float* ap2  = HAS_ACT2 ? act2 + (size_t)b * CIN_T * NSP + n0 + rc : nullptr;

    v8f acc = {};
    for (int k0 = 0; k0 < CIN_T; k0 += 32) {
        v16bf a = load_frag_row(wrow + k0, hv);
        v16bf bf;
#pragma unroll
        for (int j = 0; j < 16; ++j) {
            int k = k0 + kfrag(j, hv);
            float av = ap[(size_t)k * NSP];
            if constexpr (HAS_ACT2) av += ap2[(size_t)k * NSP];
            if constexpr (HAS_DW)   av = av * dwS[k] + dwB[k];
            bf[j] = (__bf16)av;
        }
        acc = wmma_bf16(a, bf, acc);
    }
#pragma unroll
    for (int j = 0; j < 8; ++j) {
        int co = co0 + hv * 8 + j;
        float v = acc[j];
        if constexpr (HAS_PWB) v += pwb[co];
        if constexpr (HAS_BN)  v = v * bnA[co] + bnB[co];
        if constexpr (RELU)    v = fmaxf(v, 0.f);
        out[((size_t)b * Cout + co) * NSP + n0 + rc] = v;
    }
}

// ---------------------------------------------------------------------------
// Flash-style PAM: one wave per (batch, 16-row block of energy).
// energy strip (16 x 2304) lives entirely in LDS; softmax in place;
// then sa[c,n] = gamma * sum_m V[c,m]*attn[n,m] + f1[c,n].
// ---------------------------------------------------------------------------
__global__ void __launch_bounds__(32)
k_pam(const float* __restrict__ q, const float* __restrict__ kk,
      const float* __restrict__ v, const float* __restrict__ f1,
      const float* __restrict__ gamma, float* __restrict__ sa) {
    extern __shared__ float en[];             // 16 * NSP floats = 147456 B
    int b = blockIdx.x, nb = blockIdx.y;
    int lane = threadIdx.x & 31;
    int hv = lane >> 4, rc = lane & 15;
    int n0 = nb * 16;

    // A-frag: rows = 16 output positions n, K = 16 channels padded to 32
    v16bf aq;
#pragma unroll
    for (int j = 0; j < 16; ++j) {
        int c = kfrag(j, hv);
        aq[j] = (c < 16) ? (__bf16)q[((size_t)b * 16 + c) * NSP + n0 + rc] : (__bf16)0.f;
    }
    // energy[n, m] = sum_c q[c,n] * k[c,m]
    for (int mb = 0; mb < NSP / 16; ++mb) {
        v16bf bk;
#pragma unroll
        for (int j = 0; j < 16; ++j) {
            int c = kfrag(j, hv);
            bk[j] = (c < 16) ? (__bf16)kk[((size_t)b * 16 + c) * NSP + mb * 16 + rc]
                             : (__bf16)0.f;
        }
        v8f e = {};
        e = wmma_bf16(aq, bk, e);
#pragma unroll
        for (int j = 0; j < 8; ++j)
            en[(hv * 8 + j) * NSP + mb * 16 + rc] = e[j];
    }
    __syncthreads();

    // softmax over m: row = rc, the two wave-halves split the 2304 columns
    {
        int row = rc;
        int m0 = hv * (NSP / 2), m1 = m0 + NSP / 2;
        float mx = -3.4e38f;
        for (int m = m0; m < m1; ++m) mx = fmaxf(mx, en[row * NSP + m]);
        mx = fmaxf(mx, __shfl_xor(mx, 16, 32));
        float sum = 0.f;
        for (int m = m0; m < m1; ++m) {
            float e = __expf(en[row * NSP + m] - mx);
            en[row * NSP + m] = e;
            sum += e;
        }
        sum += __shfl_xor(sum, 16, 32);
        float inv = 1.f / sum;
        for (int m = m0; m < m1; ++m) en[row * NSP + m] *= inv;
    }
    __syncthreads();

    float g = *gamma;
    // sa[c, n] = g * sum_m V[c,m] * attn[n,m] + f1[c,n]
    const float* attnRow = en + (size_t)rc * NSP;   // B-frag source (LDS, contiguous)
    for (int ct = 0; ct < 8; ++ct) {
        const float* vrow = v + ((size_t)b * INTER + ct * 16 + rc) * NSP;
        v8f acc = {};
        for (int mb = 0; mb < NSP; mb += 32) {
            v16bf av   = load_frag_row(vrow + mb, hv);      // global b128 loads
            v16bf batt = load_frag_row(attnRow + mb, hv);   // LDS b128 loads
            acc = wmma_bf16(av, batt, acc);
        }
#pragma unroll
        for (int j = 0; j < 8; ++j) {
            int c = ct * 16 + hv * 8 + j;
            size_t o = ((size_t)b * INTER + c) * NSP + n0 + rc;
            sa[o] = g * acc[j] + f1[o];
        }
    }
}

// ---------------------------------------------------------------------------
// CAM Gram: cen[b,ci,cj] = sum_n f2[ci,n]*f2[cj,n]  (K = 2304)
// Both fragments are contiguous rows => all b128 loads.
// ---------------------------------------------------------------------------
__global__ void __launch_bounds__(32)
k_cam_gram(const float* __restrict__ f2, float* __restrict__ cen) {
    int b = blockIdx.x, it = blockIdx.y, jt = blockIdx.z;
    int lane = threadIdx.x & 31, hv = lane >> 4, rc = lane & 15;
    const float* arow = f2 + ((size_t)b * INTER + it * 16 + rc) * NSP;
    const float* brow = f2 + ((size_t)b * INTER + jt * 16 + rc) * NSP;
    v8f acc = {};
    for (int mb = 0; mb < NSP; mb += 32) {
        v16bf a  = load_frag_row(arow + mb, hv);
        v16bf bb = load_frag_row(brow + mb, hv);
        acc = wmma_bf16(a, bb, acc);
    }
#pragma unroll
    for (int j = 0; j < 8; ++j)
        cen[(size_t)b * INTER * INTER + (it * 16 + hv * 8 + j) * INTER + jt * 16 + rc] = acc[j];
}

// cen -> softmax(max - cen) per row (128 threads = 128 rows)
__global__ void k_cam_softmax(const float* __restrict__ cen, float* __restrict__ cattn) {
    int b = blockIdx.x, c = threadIdx.x;
    const float* row = cen + (size_t)b * INTER * INTER + (size_t)c * INTER;
    float* o = cattn + (size_t)b * INTER * INTER + (size_t)c * INTER;
    float mx = -3.4e38f, mn = 3.4e38f;
    for (int d = 0; d < INTER; ++d) { float r = row[d]; mx = fmaxf(mx, r); mn = fminf(mn, r); }
    float vmax = mx - mn;   // max of (mx - row[d])
    float sum = 0.f;
    for (int d = 0; d < INTER; ++d) { float e = __expf((mx - row[d]) - vmax); o[d] = e; sum += e; }
    float inv = 1.f / sum;
    for (int d = 0; d < INTER; ++d) o[d] *= inv;
}

// sc[c,n] = gamma * sum_d cattn[c,d]*f2[d,n] + f2[c,n]  (K = 128)
__global__ void __launch_bounds__(32)
k_cam_apply(const float* __restrict__ cattn, const float* __restrict__ f2,
            const float* __restrict__ gamma, float* __restrict__ sc) {
    int b = blockIdx.x, ct = blockIdx.y, nb = blockIdx.z;
    int lane = threadIdx.x & 31, hv = lane >> 4, rc = lane & 15;
    int n0 = nb * 16;
    const float* arow = cattn + (size_t)b * INTER * INTER + (size_t)(ct * 16 + rc) * INTER;
    const float* bp   = f2 + (size_t)b * INTER * NSP + n0 + rc;   // stride NSP over d
    v8f acc = {};
#pragma unroll
    for (int k0 = 0; k0 < INTER; k0 += 32) {
        v16bf a = load_frag_row(arow + k0, hv);   // contiguous -> b128
        v16bf bb;
#pragma unroll
        for (int j = 0; j < 16; ++j) {
            int d = k0 + kfrag(j, hv);
            bb[j] = (__bf16)bp[(size_t)d * NSP];
        }
        acc = wmma_bf16(a, bb, acc);
    }
    float g = *gamma;
#pragma unroll
    for (int j = 0; j < 8; ++j) {
        int c = ct * 16 + hv * 8 + j;
        size_t o = ((size_t)b * INTER + c) * NSP + n0 + rc;
        sc[o] = g * acc[j] + f2[o];
    }
}

// ---------------------------------------------------------------------------
extern "C" void kernel_launch(void* const* d_in, const int* in_sizes, int n_in,
                              void* d_out, int out_size, void* d_ws, size_t ws_size,
                              hipStream_t stream) {
    (void)in_sizes; (void)n_in; (void)out_size; (void)ws_size;
    // setup_inputs() flattening order
    const float* X       = (const float*)d_in[0];
    const float* c5a_dw  = (const float*)d_in[1];
    const float* c5a_pw  = (const float*)d_in[2];
    const float* bn5a_s  = (const float*)d_in[3];
    const float* bn5a_b  = (const float*)d_in[4];
    const float* bn5a_m  = (const float*)d_in[5];
    const float* bn5a_v  = (const float*)d_in[6];
    const float* c5c_dw  = (const float*)d_in[7];
    const float* c5c_pw  = (const float*)d_in[8];
    const float* bn5c_s  = (const float*)d_in[9];
    const float* bn5c_b  = (const float*)d_in[10];
    const float* bn5c_m  = (const float*)d_in[11];
    const float* bn5c_v  = (const float*)d_in[12];
    const float* q_dwW   = (const float*)d_in[13];
    const float* q_pwW   = (const float*)d_in[14];
    const float* q_dwB   = (const float*)d_in[15];
    const float* q_pwB   = (const float*)d_in[16];
    const float* kq_dwW  = (const float*)d_in[17];
    const float* kq_pwW  = (const float*)d_in[18];
    const float* kq_dwB  = (const float*)d_in[19];
    const float* kq_pwB  = (const float*)d_in[20];
    const float* v_dwW   = (const float*)d_in[21];
    const float* v_pwW   = (const float*)d_in[22];
    const float* v_dwB   = (const float*)d_in[23];
    const float* v_pwB   = (const float*)d_in[24];
    const float* g_pam   = (const float*)d_in[25];
    const float* g_cam   = (const float*)d_in[26];
    const float* c51_dw  = (const float*)d_in[27];
    const float* c51_pw  = (const float*)d_in[28];
    const float* bn51_s  = (const float*)d_in[29];
    const float* bn51_b  = (const float*)d_in[30];
    const float* bn51_m  = (const float*)d_in[31];
    const float* bn51_v  = (const float*)d_in[32];
    const float* c52_dw  = (const float*)d_in[33];
    const float* c52_pw  = (const float*)d_in[34];
    const float* bn52_s  = (const float*)d_in[35];
    const float* bn52_b  = (const float*)d_in[36];
    const float* bn52_m  = (const float*)d_in[37];
    const float* bn52_v  = (const float*)d_in[38];
    const float* c6_dw   = (const float*)d_in[39];
    const float* c6_pw   = (const float*)d_in[40];
    const float* c6_db   = (const float*)d_in[41];
    const float* c6_pb   = (const float*)d_in[42];

    float* ws = (float*)d_ws;
    const size_t PLANE512 = (size_t)BATCH * CIN * NSP;     // 18,874,368
    const size_t PLANE128 = (size_t)BATCH * INTER * NSP;   //  4,718,592
    const size_t PLANE16  = (size_t)BATCH * 16 * NSP;
    float* dwbuf = ws;                 size_t off = PLANE512;
    float* f1    = ws + off;           off += PLANE128;
    float* f2    = ws + off;           off += PLANE128;
    float* vbuf  = ws + off;           off += PLANE128;
    float* sa    = ws + off;           off += PLANE128;
    float* sa2   = ws + off;           off += PLANE128;
    float* sc    = ws + off;           off += PLANE128;
    float* sc2   = ws + off;           off += PLANE128;
    float* t128  = ws + off;           off += PLANE128;
    float* qbuf  = ws + off;           off += PLANE16;
    float* kbuf  = ws + off;           off += PLANE16;
    float* cen   = ws + off;           off += (size_t)BATCH * INTER * INTER;
    float* cattn = ws + off;           off += (size_t)BATCH * INTER * INTER;
    float* bnA   = ws + off;           off += 512;
    float* bnB   = ws + off;           off += 512;

    const int NT = NSP / 16;  // 144 spatial tiles
    dim3 w32(32);

    // ---- PAM branch ----
    k_bnprep<<<1, 128, 0, stream>>>(bn5a_s, bn5a_b, bn5a_m, bn5a_v, bnA, bnB, INTER);
    {
        int total = BATCH * CIN * NSP;
        k_dw3x3<<<(total + 255) / 256, 256, 0, stream>>>(X, c5a_dw, dwbuf, CIN);
    }
    k_pwgemm<512, false, false, false, true, true>
        <<<dim3(BATCH, INTER / 16, NT), w32, 0, stream>>>(
        dwbuf, nullptr, c5a_pw, nullptr, nullptr, nullptr, bnA, bnB, f1, INTER);

    k_pwgemm<128, false, true, true, false, false>
        <<<dim3(BATCH, 1, NT), w32, 0, stream>>>(
        f1, nullptr, q_pwW, q_dwW, q_dwB, q_pwB, nullptr, nullptr, qbuf, 16);
    k_pwgemm<128, false, true, true, false, false>
        <<<dim3(BATCH, 1, NT), w32, 0, stream>>>(
        f1, nullptr, kq_pwW, kq_dwW, kq_dwB, kq_pwB, nullptr, nullptr, kbuf, 16);
    k_pwgemm<128, false, true, true, false, false>
        <<<dim3(BATCH, INTER / 16, NT), w32, 0, stream>>>(
        f1, nullptr, v_pwW, v_dwW, v_dwB, v_pwB, nullptr, nullptr, vbuf, INTER);

    size_t pam_lds = (size_t)16 * NSP * sizeof(float);     // 147456 B
    k_pam<<<dim3(BATCH, NT), w32, pam_lds, stream>>>(qbuf, kbuf, vbuf, f1, g_pam, sa);

    k_bnprep<<<1, 128, 0, stream>>>(bn51_s, bn51_b, bn51_m, bn51_v, bnA, bnB, INTER);
    {
        int total = BATCH * INTER * NSP;
        k_dw3x3<<<(total + 255) / 256, 256, 0, stream>>>(sa, c51_dw, t128, INTER);
    }
    k_pwgemm<128, false, false, false, true, true>
        <<<dim3(BATCH, INTER / 16, NT), w32, 0, stream>>>(
        t128, nullptr, c51_pw, nullptr, nullptr, nullptr, bnA, bnB, sa2, INTER);

    // ---- CAM branch ----
    k_bnprep<<<1, 128, 0, stream>>>(bn5c_s, bn5c_b, bn5c_m, bn5c_v, bnA, bnB, INTER);
    {
        int total = BATCH * CIN * NSP;
        k_dw3x3<<<(total + 255) / 256, 256, 0, stream>>>(X, c5c_dw, dwbuf, CIN);
    }
    k_pwgemm<512, false, false, false, true, true>
        <<<dim3(BATCH, INTER / 16, NT), w32, 0, stream>>>(
        dwbuf, nullptr, c5c_pw, nullptr, nullptr, nullptr, bnA, bnB, f2, INTER);

    k_cam_gram<<<dim3(BATCH, INTER / 16, INTER / 16), w32, 0, stream>>>(f2, cen);
    k_cam_softmax<<<BATCH, 128, 0, stream>>>(cen, cattn);
    k_cam_apply<<<dim3(BATCH, INTER / 16, NT), w32, 0, stream>>>(cattn, f2, g_cam, sc);

    k_bnprep<<<1, 128, 0, stream>>>(bn52_s, bn52_b, bn52_m, bn52_v, bnA, bnB, INTER);
    {
        int total = BATCH * INTER * NSP;
        k_dw3x3<<<(total + 255) / 256, 256, 0, stream>>>(sc, c52_dw, t128, INTER);
    }
    k_pwgemm<128, false, false, false, true, true>
        <<<dim3(BATCH, INTER / 16, NT), w32, 0, stream>>>(
        t128, nullptr, c52_pw, nullptr, nullptr, nullptr, bnA, bnB, sc2, INTER);

    // ---- fuse: conv6((sa2 + sc2)), dw-1x1 + pw 128->512, bias, no BN/ReLU ----
    k_pwgemm<128, true, true, true, false, false>
        <<<dim3(BATCH, COUTC / 16, NT), w32, 0, stream>>>(
        sa2, sc2, c6_pw, c6_dw, c6_db, c6_pb, nullptr, nullptr,
        (float*)d_out, COUTC);
}
